// FourierFilterBanks_42923903156651
// MI455X (gfx1250) — matrix operand order repaired and checked by reading, not compiled, not run on hardware
//
#include <hip/hip_runtime.h>
#include <hip/hip_bf16.h>
#include <math.h>

typedef __attribute__((ext_vector_type(16))) _Float16 v16h;
typedef __attribute__((ext_vector_type(8)))  _Float16 v8h;
typedef __attribute__((ext_vector_type(8)))  float    v8f;
typedef __attribute__((ext_vector_type(4)))  unsigned int u32x4;
typedef __attribute__((ext_vector_type(8)))  int i32x8;
typedef __attribute__((ext_vector_type(4)))  int i32x4;

#define N_PTS_       (1 << 20)
#define T_           (1 << 19)
#define NLV          8
#define BLK          256
#define PTS_PER_BLK  128

// per-level packed weight blob in d_ws (units: _Float16 slots)
#define WSBLK        8192            // 16 KB per level
#define OF_WT        0               // Ws[i-1]^T f16 [64][64]  (level 0: [W0^T|b0|0...] in K=0..3)
#define OF_BT        4096            // Bmat[i]^T f16 [32][64]
#define OF_GW        6144            // Gw[i] f16 [2][64]
#define OF_GB        6272            // Gb[i] f16 [64]
#define OF_BIAS      6336            // bs[i-1] f32 [64] (level 0: zeros)
#define LVL_HALVES   6464            // copied halves per level (12928 B)
#define LVL_BYTES    (LVL_HALVES * 2)
#define OF_HEAD      (8 * WSBLK)     // Wout^T f16 [16][64] + bout f32 [16]
#define HEAD_HALVES  1056            // 1024 + 32 (16 floats)

__device__ __constant__ float c_res[8] = {16.f, 26.f, 43.f, 70.f, 115.f, 190.f, 312.f, 512.f};

struct HPair { v8h lo, hi; };
union AF { v16h v; HPair p; };

// D = A*B + C ; A = weights (lane=row), B = activations (lane=point)
#define WMMA_T(A, B, C, RB) \
    __builtin_amdgcn_wmma_f32_16x16x32_f16(false, (A), false, (B), (short)0, (C), false, (RB))

// lane <-> lane^16 exchange on the VALU pipe (v_permlanex16_b32, identity sel)
static __device__ __forceinline__ int lane_xor16(int v)
{
#if __has_builtin(__builtin_amdgcn_permlanex16)
    return __builtin_amdgcn_permlanex16(v, v, 0x76543210, 0xfedcba98, false, false);
#else
    return __shfl_xor(v, 16);
#endif
}

// Convert 4 WMMA D tiles (C-layout: lane=point, VGPR r = channel mt*16+r+8*kg)
// into 2 B-operand fragments (lane=point, halves = channel kt*32+kg*16+j).
// Within-lane f16 pack + one xor-16 lane exchange; no LDS.
static __device__ __forceinline__ void d_to_b(const v8f* d, int kg, AF* b)
{
    union PH { v8h h; int w[4]; };
    PH own[4], oth[4];
    #pragma unroll
    for (int mt = 0; mt < 4; ++mt) {
        #pragma unroll
        for (int j = 0; j < 8; ++j) own[mt].h[j] = (_Float16)d[mt][j];
        #pragma unroll
        for (int q = 0; q < 4; ++q) oth[mt].w[q] = lane_xor16(own[mt].w[q]);
    }
    #pragma unroll
    for (int kt = 0; kt < 2; ++kt) {
        int mt = kt * 2 + kg;
        b[kt].p.lo = kg ? oth[mt].h : own[mt].h;
        b[kt].p.hi = kg ? own[mt].h : oth[mt].h;
    }
}

#if __has_builtin(__builtin_amdgcn_tensor_load_to_lds)
#define HAVE_TDM 1
// async 1-D TDM copy of one level blob (12.9 KB) into LDS; tracked by TENSORcnt
static __device__ __forceinline__ void tdm_load_lvl(const _Float16* gsrc, const _Float16* ldst)
{
    uint64_t ga = (uint64_t)(uintptr_t)gsrc;
    unsigned la = (unsigned)(uintptr_t)(const void*)ldst;   // flat addr[31:0] == LDS offset
    const unsigned NE = LVL_BYTES / 8;                      // 1616 8-byte elements
    u32x4 g0 = { 1u,                                        // count=1 (valid user D#)
                 la,                                        // lds_addr
                 (unsigned)ga,
                 (unsigned)((ga >> 32) & 0x01ffffffu) | 0x80000000u }; // type=2
    i32x8 g1 = { (int)0x00030000u,                          // data_size=3 (8B), no multicast
                 (int)(NE << 16),                           // tensor_dim0[15:0]
                 (int)(1u << 16),                           // tensor_dim1=1
                 (int)(NE << 16),                           // tile_dim0
                 0,                                         // tile_dim1/2 unused (1-D)
                 (int)NE,                                   // tensor_dim0_stride low32
                 0, 0 };
    i32x4 gz4 = { 0, 0, 0, 0 };
    i32x8 gz8 = { 0, 0, 0, 0, 0, 0, 0, 0 };
    __builtin_amdgcn_tensor_load_to_lds(g0, g1, gz4, gz4, gz8, 0);
}
#else
#define HAVE_TDM 0
#endif

// ---- prep: pack/transposed-convert all weights into d_ws, once per launch ----
__global__ __launch_bounds__(BLK) void ffb_prep(
    const float* __restrict__ W0,   const float* __restrict__ b0,
    const float* __restrict__ Ws,   const float* __restrict__ bs,
    const float* __restrict__ Gw,   const float* __restrict__ Gb,
    const float* __restrict__ Bmat, const float* __restrict__ Wout,
    const float* __restrict__ bout, _Float16* __restrict__ wsh)
{
    const int b = blockIdx.x, t = threadIdx.x;
    if (b < NLV) {
        _Float16* base = wsh + (size_t)b * WSBLK;
        if (b == 0) {
            // full 64x64 tile: [W0^T | b0 | 0...] -> level 0 shares the generic path
            for (int idx = t; idx < 4096; idx += BLK) {
                int n = idx >> 6, k = idx & 63;
                float f = (k < 3) ? W0[k * 64 + n] : (k == 3 ? b0[n] : 0.f);
                base[OF_WT + idx] = (_Float16)f;
            }
        } else {
            const float* W = Ws + (size_t)(b - 1) * 4096;
            for (int idx = t; idx < 4096; idx += BLK) {
                int k = idx >> 6, n = idx & 63;
                base[OF_WT + n * 64 + k] = (_Float16)W[idx];
            }
        }
        const float* Bm = Bmat + (size_t)b * 2048;
        for (int idx = t; idx < 2048; idx += BLK) {
            int k = idx >> 5, j = idx & 31;
            base[OF_BT + j * 64 + k] = (_Float16)Bm[idx];
        }
        for (int idx = t; idx < 128; idx += BLK) base[OF_GW + idx] = (_Float16)Gw[b * 128 + idx];
        for (int idx = t; idx < 64; idx += BLK)  base[OF_GB + idx] = (_Float16)Gb[b * 64 + idx];
        float* bias = (float*)(base + OF_BIAS);
        for (int idx = t; idx < 64; idx += BLK)  bias[idx] = (b > 0) ? bs[(b - 1) * 64 + idx] : 0.f;
    } else {
        _Float16* hb = wsh + OF_HEAD;
        for (int idx = t; idx < 1024; idx += BLK) {
            int n = idx >> 6, k = idx & 63;
            hb[n * 64 + k] = (_Float16)Wout[k * 16 + n];
        }
        float* bo = (float*)(hb + 1024);
        for (int idx = t; idx < 16; idx += BLK) bo[idx] = bout[idx];
    }
}

__global__ __launch_bounds__(BLK) void ffb_kernel(
    const float* __restrict__ input, const float* __restrict__ tables,
    const _Float16* __restrict__ wsh, float* __restrict__ out)
{
    __shared__ alignas(16) _Float16 sLvl[2][LVL_HALVES];  // double-buffered TDM tiles
    __shared__ alignas(16) _Float16 sHead[HEAD_HALVES];   // Wout^T + bout

    const _Float16* sWoutT = sHead;
    const float*    sBout  = (const float*)(sHead + 1024);

    const int tid  = threadIdx.x;
    const int wave = tid >> 5;
    const int lane = tid & 31;
    const int lp   = lane & 15;   // point within wave tile (B/D column)
    const int kg   = lane >> 4;   // K-half / row-half group
    const int pbase = blockIdx.x * PTS_PER_BLK + wave * 16;
    const int pidx  = pbase + lp;

    // ---- one-time head staging (2.1 KB, plain vector copy) ----
    {
        const uint32_t* src = (const uint32_t*)(wsh + OF_HEAD);
        uint32_t* dst = (uint32_t*)sHead;
        for (int idx = tid; idx < HEAD_HALVES / 2; idx += BLK) dst[idx] = src[idx];
    }

    // ---- per-point prologue (each lane owns point lp) ----
    float x0 = input[pidx * 3 + 0] * (1.0f / 1.5f);
    float x1 = input[pidx * 3 + 1] * (1.0f / 1.5f);
    float x2 = input[pidx * 3 + 2] * (1.0f / 1.5f);
    float xn0 = fminf(fmaxf((x0 + 1.f) * 0.5f, 0.f), 1.f - 1e-6f);
    float xn1 = fminf(fmaxf((x1 + 1.f) * 0.5f, 0.f), 1.f - 1e-6f);
    float xn2 = fminf(fmaxf((x2 + 1.f) * 0.5f, 0.f), 1.f - 1e-6f);

    // seed activations with the input embedding: B = [x0,x1,x2,1, 0...]
    // level 0's A tile is [W0^T|b0|0...], so the generic loop computes x@W0+b0.
    AF hB[2];
    hB[0].v = v16h{};
    hB[1].v = v16h{};
    if (kg == 0) {
        hB[0].v[0] = (_Float16)x0; hB[0].v[1] = (_Float16)x1;
        hB[0].v[2] = (_Float16)x2; hB[0].v[3] = (_Float16)1.0f;
    }

    v8f acc[4];
    #pragma unroll
    for (int mt = 0; mt < 4; ++mt)
        for (int r = 0; r < 8; ++r) acc[mt][r] = 0.f;

    // ---- level-0 blob: DMA in before the loop ----
#if HAVE_TDM
    if (wave == 0) {
        tdm_load_lvl(wsh, sLvl[0]);
        __builtin_amdgcn_s_wait_tensorcnt(0);
    }
#else
    {
        const uint4* src = (const uint4*)wsh;
        uint4* dst = (uint4*)sLvl[0];
        for (int idx = tid; idx < LVL_BYTES / 16; idx += BLK) dst[idx] = src[idx];
    }
#endif
    __syncthreads();

    #pragma unroll 1
    for (int i = 0; i < NLV; ++i) {
        const _Float16* cur  = sLvl[i & 1];
        const _Float16* sWt  = cur + OF_WT;
        const _Float16* sBt  = cur + OF_BT;
        const _Float16* sGwh = cur + OF_GW;
        const _Float16* sGbh = cur + OF_GB;
        const float*    sBias = (const float*)(cur + OF_BIAS);

#if HAVE_TDM
        // pipeline: kick next level's DMA into the other buffer, overlap with compute
        if (i + 1 < NLV && wave == 0)
            tdm_load_lvl(wsh + (size_t)(i + 1) * WSBLK, sLvl[(i + 1) & 1]);
#endif
        if (i + 2 < NLV && tid == 0)
            __builtin_prefetch(wsh + (size_t)(i + 2) * WSBLK, 0, 1);

        // ---- hash-grid trilinear lookup (2 lanes/point x 4 corners) ----
        float g0s = 0.f, g1s = 0.f;
        {
            float res = c_res[i];
            float px = xn0 * res, py = xn1 * res, pz = xn2 * res;
            float f0 = floorf(px), f1 = floorf(py), f2 = floorf(pz);
            float w0 = px - f0, w1 = py - f1, w2 = pz - f2;
            unsigned u0 = (unsigned)f0, u1 = (unsigned)f1, u2 = (unsigned)f2;
            const float2* tb = (const float2*)tables + (size_t)i * T_;
            #pragma unroll
            for (int j = 0; j < 4; ++j) {
                int bx = kg, by = (j >> 1) & 1, bz = j & 1;
                unsigned ha = u0 + (unsigned)bx;                       // prime 1
                unsigned hb = (u1 + (unsigned)by) * 2654435761u;
                unsigned hc = (u2 + (unsigned)bz) * 805459861u;
                unsigned id = (ha ^ hb ^ hc) & (unsigned)(T_ - 1);
                float2 f = tb[id];                                     // L2-resident gather
                float wc = (bx ? w0 : 1.f - w0) * (by ? w1 : 1.f - w1) * (bz ? w2 : 1.f - w2);
                g0s += wc * f.x;
                g1s += wc * f.y;
            }
            g0s += __int_as_float(lane_xor16(__float_as_int(g0s)));  // all lanes: g(point lp)
            g1s += __int_as_float(lane_xor16(__float_as_int(g1s)));
        }

        // ---- h update: h^T_new = W^T @ h^T (weights in A, activations in B) ----
        v8f d[4];
        // bias folded into accumulator init (per-channel, same for all lanes)
        #pragma unroll
        for (int mt = 0; mt < 4; ++mt) {
            float4 blo = *reinterpret_cast<const float4*>(sBias + mt * 16 + kg * 8);
            float4 bhi = *reinterpret_cast<const float4*>(sBias + mt * 16 + kg * 8 + 4);
            d[mt][0] = blo.x; d[mt][1] = blo.y; d[mt][2] = blo.z; d[mt][3] = blo.w;
            d[mt][4] = bhi.x; d[mt][5] = bhi.y; d[mt][6] = bhi.z; d[mt][7] = bhi.w;
        }
        #pragma unroll
        for (int kt = 0; kt < 2; ++kt) {
            AF aw[4];
            #pragma unroll
            for (int mt = 0; mt < 4; ++mt) {
                int rb = (mt * 16 + lp) * 64 + kt * 32 + kg * 16;
                aw[mt].p.lo = *reinterpret_cast<const v8h*>(sWt + rb);
                aw[mt].p.hi = *reinterpret_cast<const v8h*>(sWt + rb + 8);
            }
            d[0] = WMMA_T(aw[0].v, hB[kt].v, d[0], true);
            d[1] = WMMA_T(aw[1].v, hB[kt].v, d[1], true);
            d[2] = WMMA_T(aw[2].v, hB[kt].v, d[2], true);
            d[3] = WMMA_T(aw[3].v, hB[kt].v, d[3], false);
        }

        // ---- D -> B fragments for next level (in-register transpose) ----
        d_to_b(d, kg, hB);

        // ---- hg = h + g@Gw + Gb, packed f16; then Fourier proj WMMA ----
        {
            _Float16 g0h = (_Float16)g0s, g1h = (_Float16)g1s;
            v16h g0v = {g0h,g0h,g0h,g0h,g0h,g0h,g0h,g0h,g0h,g0h,g0h,g0h,g0h,g0h,g0h,g0h};
            v16h g1v = {g1h,g1h,g1h,g1h,g1h,g1h,g1h,g1h,g1h,g1h,g1h,g1h,g1h,g1h,g1h,g1h};
            AF hg[2];
            #pragma unroll
            for (int kt = 0; kt < 2; ++kt) {
                AF gw0f, gw1f, gbf;
                int ko = kt * 32 + kg * 16;
                gw0f.p.lo = *reinterpret_cast<const v8h*>(sGwh + ko);
                gw0f.p.hi = *reinterpret_cast<const v8h*>(sGwh + ko + 8);
                gw1f.p.lo = *reinterpret_cast<const v8h*>(sGwh + 64 + ko);
                gw1f.p.hi = *reinterpret_cast<const v8h*>(sGwh + 64 + ko + 8);
                gbf.p.lo  = *reinterpret_cast<const v8h*>(sGbh + ko);
                gbf.p.hi  = *reinterpret_cast<const v8h*>(sGbh + ko + 8);
                hg[kt].v = hB[kt].v + g0v * gw0f.v + g1v * gw1f.v + gbf.v;  // v_pk_fma_f16
            }
            v8f dp[2];
            #pragma unroll
            for (int m2 = 0; m2 < 2; ++m2)
                for (int r = 0; r < 8; ++r) dp[m2][r] = 0.f;
            #pragma unroll
            for (int kt = 0; kt < 2; ++kt) {
                AF ab[2];
                #pragma unroll
                for (int m2 = 0; m2 < 2; ++m2) {
                    int rb = (m2 * 16 + lp) * 64 + kt * 32 + kg * 16;
                    ab[m2].p.lo = *reinterpret_cast<const v8h*>(sBt + rb);
                    ab[m2].p.hi = *reinterpret_cast<const v8h*>(sBt + rb + 8);
                }
                dp[0] = WMMA_T(ab[0].v, hg[kt].v, dp[0], true);
                dp[1] = WMMA_T(ab[1].v, hg[kt].v, dp[1], false);
            }
            // fourier: acc[0:32] += sin(2*pi*proj), acc[32:64] += cos(2*pi*proj)
            // HW V_SIN/COS_F32 take revolutions, so feed proj directly.
            #pragma unroll
            for (int m2 = 0; m2 < 2; ++m2) {
                #pragma unroll
                for (int r = 0; r < 8; ++r) {
                    float p = dp[m2][r];
                    acc[m2][r]     += __builtin_amdgcn_sinf(p);
                    acc[m2 + 2][r] += __builtin_amdgcn_cosf(p);
                }
            }
        }

        // ---- drain next level's DMA, release this level's buffer ----
#if HAVE_TDM
        if (wave == 0 && i + 1 < NLV)
            __builtin_amdgcn_s_wait_tensorcnt(0);
        __syncthreads();
#else
        __syncthreads();   // all reads of buffers done
        if (i + 1 < NLV) {
            const uint4* src = (const uint4*)(wsh + (size_t)(i + 1) * WSBLK);
            uint4* dst = (uint4*)sLvl[(i + 1) & 1];
            for (int idx = tid; idx < LVL_BYTES / 16; idx += BLK) dst[idx] = src[idx];
            __syncthreads();
        }
#endif
    }

    // ---- output head: out^T = Wout^T @ acc^T + bout ----
    {
        AF aB[2];
        d_to_b(acc, kg, aB);
        v8f c;
        float4 blo = *reinterpret_cast<const float4*>(sBout + kg * 8);
        float4 bhi = *reinterpret_cast<const float4*>(sBout + kg * 8 + 4);
        c[0] = blo.x; c[1] = blo.y; c[2] = blo.z; c[3] = blo.w;
        c[4] = bhi.x; c[5] = bhi.y; c[6] = bhi.z; c[7] = bhi.w;
        #pragma unroll
        for (int kt = 0; kt < 2; ++kt) {
            AF aw;
            int rb = lp * 64 + kt * 32 + kg * 16;
            aw.p.lo = *reinterpret_cast<const v8h*>(sWoutT + rb);
            aw.p.hi = *reinterpret_cast<const v8h*>(sWoutT + rb + 8);
            c = WMMA_T(aw.v, aB[kt].v, c, false);
        }
        // lane owns point lp, channels r+8*kg: two coalesced 16B stores
        float4 o0 = make_float4(c[0], c[1], c[2], c[3]);
        float4 o1 = make_float4(c[4], c[5], c[6], c[7]);
        float* op = out + (size_t)(pbase + lp) * 16 + kg * 8;
        *reinterpret_cast<float4*>(op)     = o0;
        *reinterpret_cast<float4*>(op + 4) = o1;
    }
}

extern "C" void kernel_launch(void* const* d_in, const int* in_sizes, int n_in,
                              void* d_out, int out_size, void* d_ws, size_t ws_size,
                              hipStream_t stream) {
    (void)in_sizes; (void)n_in; (void)out_size; (void)ws_size;
    const float* input  = (const float*)d_in[0];
    const float* W0     = (const float*)d_in[1];
    const float* b0     = (const float*)d_in[2];
    const float* Ws     = (const float*)d_in[3];
    const float* bs     = (const float*)d_in[4];
    const float* tables = (const float*)d_in[5];
    const float* Gw     = (const float*)d_in[6];
    const float* Gb     = (const float*)d_in[7];
    const float* Bmat   = (const float*)d_in[8];
    const float* Wout   = (const float*)d_in[9];
    const float* bout   = (const float*)d_in[10];
    float* out = (float*)d_out;
    _Float16* wsh = (_Float16*)d_ws;

    hipLaunchKernelGGL(ffb_prep, dim3(NLV + 1), dim3(BLK), 0, stream,
                       W0, b0, Ws, bs, Gw, Gb, Bmat, Wout, bout, wsh);
    hipLaunchKernelGGL(ffb_kernel, dim3(N_PTS_ / PTS_PER_BLK), dim3(BLK), 0, stream,
                       input, tables, wsh, out);
}
